// Attention_79680233276235
// MI455X (gfx1250) — matrix operand order repaired and checked by reading, not compiled, not run on hardware
//
#include <hip/hip_runtime.h>
#include <hip/hip_bf16.h>

// ---------------- problem constants ----------------
#define BB   2
#define SS   2048
#define HH   4096
#define NH_  32
#define NKV_ 8
#define HD_  128
#define GR_  (NH_ / NKV_)   // 4

// ---------------- vector types ----------------
typedef __attribute__((ext_vector_type(16))) __bf16 v16bf;
typedef __attribute__((ext_vector_type(8)))  __bf16 v8bf;
typedef __attribute__((ext_vector_type(4)))  __bf16 v4bf;
typedef __attribute__((ext_vector_type(8)))  float  v8f;
typedef __attribute__((ext_vector_type(4)))  float  v4f;
typedef __attribute__((ext_vector_type(4)))  unsigned int v4u;
typedef __attribute__((ext_vector_type(8)))  int    v8i;
typedef __attribute__((ext_vector_type(4)))  int    v4i;

#if defined(__has_builtin)
#if __has_builtin(__builtin_amdgcn_tensor_load_to_lds) && __has_builtin(__builtin_amdgcn_s_wait_tensorcnt)
#define HAVE_TDM 1
#endif
#endif

// ---------------------------------------------------------------------------
// Fragment loaders (CDNA5 WMMA VGPR layouts, wave32)
// A (16x32 bf16): row M = lane&15 ; k = (j&7) + (j>=8)*16 + (lane>=16)*8
// B (32x16 bf16, column-major in LDS [n][k]): col N = lane&15 ; k = (j&7) + (j>=8)*16 + (lane>=16)*8
// ---------------------------------------------------------------------------
__device__ __forceinline__ v16bf load_a_frag(const __bf16* base, int strideH, int lane) {
    const int row  = lane & 15;
    const int koff = (lane >> 4) << 3;           // 0 or 8
    const __bf16* p = base + row * strideH + koff;
    v8bf lo = *(const v8bf*)(p);                 // k = koff + 0..7
    v8bf hi = *(const v8bf*)(p + 16);            // k = koff + 16..23
    v16bf a;
#pragma unroll
    for (int i = 0; i < 8; ++i) { a[i] = lo[i]; a[i + 8] = hi[i]; }
    return a;
}

__device__ __forceinline__ v16bf load_b_frag(const __bf16* base, int strideH, int lane) {
    const int col  = lane & 15;
    const int koff = (lane >> 4) << 3;           // 0 or 8
    const __bf16* p = base + col * strideH + koff;
    v8bf lo = *(const v8bf*)(p);
    v8bf hi = *(const v8bf*)(p + 16);
    v16bf b;
#pragma unroll
    for (int i = 0; i < 8; ++i) { b[i] = lo[i]; b[i + 8] = hi[i]; }
    return b;
}

__device__ __forceinline__ v8f wmma_bf16(v16bf a, v16bf b, v8f c) {
    return __builtin_amdgcn_wmma_f32_16x16x32_bf16(false, a, false, b, (short)0, c,
                                                   false, false);
}

// ---------------------------------------------------------------------------
// TDM: DMA a 2D bf16 tile [tile_h rows x tile_w elems] (row stride =
// stride_elems in global) into LDS at lds_byte_off, padding 4 DWORDs after
// every 16 DWORDs stored -> LDS row stride = tile_w(32) + 8 = 40 halves.
// D# layout per CDNA5 ISA 8.3/8.4 (group0 128b, group1 256b; groups 2/3 and
// the trailing descriptor words zero for a 2D tile).
// 6-arg builtin form: (u32x4 g0, i32x8 g1, i32x4, i32x4, i32x8, i32 cpol).
// ---------------------------------------------------------------------------
#ifdef HAVE_TDM
__device__ __forceinline__ void tdm_load_2d_bf16(unsigned lds_byte_off,
                                                 const __bf16* gsrc,
                                                 unsigned tile_w, unsigned tile_h,
                                                 unsigned stride_elems) {
    const unsigned long long ga = (unsigned long long)(const void*)gsrc;
    v4u g0;
    g0[0] = 1u;                                              // count=1, user mode
    g0[1] = lds_byte_off;                                    // lds_addr
    g0[2] = (unsigned)(ga & 0xFFFFFFFFu);                    // global_addr[31:0]
    g0[3] = (unsigned)((ga >> 32) & 0x01FFFFFFu) | (2u << 30); // addr[56:32]|type=2
    const unsigned d0 = (1u << 16)    // data_size = 2 bytes
                      | (1u << 20)    // pad_enable
                      | (3u << 22)    // pad_interval: 16 DWORDs
                      | (3u << 25);   // pad_amount:   4 DWORDs
    v8i g1;
    g1[0] = (int)d0;
    g1[1] = (int)((tile_w & 0xFFFFu) << 16);                           // tensor_dim0 lo
    g1[2] = (int)(((tile_w >> 16) & 0xFFFFu) | ((tile_h & 0xFFFFu) << 16)); // t_dim0 hi|t_dim1 lo
    g1[3] = (int)(((tile_h >> 16) & 0xFFFFu) | ((tile_w & 0xFFFFu) << 16)); // t_dim1 hi|tile_dim0
    g1[4] = (int)(tile_h & 0xFFFFu);                                   // tile_dim1 (tile_dim2=0)
    g1[5] = (int)stride_elems;                                         // dim0_stride lo32
    g1[6] = 0;
    g1[7] = 0;
    const v4i gz4 = {0, 0, 0, 0};
    const v8i gz8 = {0, 0, 0, 0, 0, 0, 0, 0};
    __builtin_amdgcn_tensor_load_to_lds(g0, g1, gz4, gz4, gz8, 0);
}
#endif

__device__ __forceinline__ unsigned lds_off_u32(const void* p) {
    // LDS aperture maps flat addr[31:0] -> LDS byte offset (ISA 10.2)
    return (unsigned)(unsigned long long)p;
}

// ---------------------------------------------------------------------------
// fp32 -> bf16 elementwise conversion (one-time, bandwidth trivial)
// ---------------------------------------------------------------------------
__global__ void f32_to_bf16_kernel(const float* __restrict__ src,
                                   __bf16* __restrict__ dst, long long n) {
    const long long i = ((long long)blockIdx.x * blockDim.x + threadIdx.x) * 8;
    if (i + 8 <= n) {
        v4f a = *(const v4f*)(src + i);
        v4f b = *(const v4f*)(src + i + 4);
        v4bf ha, hb;
#pragma unroll
        for (int j = 0; j < 4; ++j) { ha[j] = (__bf16)a[j]; hb[j] = (__bf16)b[j]; }
        *(v4bf*)(dst + i)     = ha;
        *(v4bf*)(dst + i + 4) = hb;
    }
}

// ---------------------------------------------------------------------------
// GEMM: Out[M,N] = A[M,K] * W[N,K]^T  (bf16 in, fp32 out, WMMA compute)
// Block tile 128x128, 8 waves (2 M x 4 N), wave tile 64x32. K step 32.
// Tiles staged to LDS by TDM (double-buffered, TENSORcnt-synchronized).
// ---------------------------------------------------------------------------
#define LDT 40   // LDS row stride (halves): 80B, 16B aligned, conflict-skewed

__global__ __launch_bounds__(256) void gemm_xwt_kernel(const __bf16* __restrict__ X,
                                                       const __bf16* __restrict__ W,
                                                       float* __restrict__ Out,
                                                       int M, int N, int K) {
    __shared__ __align__(16) __bf16 lsA[2][128 * LDT];
    __shared__ __align__(16) __bf16 lsB[2][128 * LDT];

    const int tid  = threadIdx.x;
    const int lane = tid & 31;
    const int wave = tid >> 5;
    const int wm   = wave & 1;
    const int wn   = wave >> 1;
    const int m0   = blockIdx.x * 128;
    const int n0   = blockIdx.y * 128;

    v8f acc[4][2];
    const v8f zf = {0.f, 0.f, 0.f, 0.f, 0.f, 0.f, 0.f, 0.f};
#pragma unroll
    for (int mt = 0; mt < 4; ++mt)
#pragma unroll
        for (int nt = 0; nt < 2; ++nt) acc[mt][nt] = zf;

    const int steps = K / 32;
    const __bf16* gA = X + (size_t)m0 * K;
    const __bf16* gB = W + (size_t)n0 * K;

#ifdef HAVE_TDM
    if (wave == 0) {   // one DMA issue per op (TDM ignores EXEC)
        tdm_load_2d_bf16(lds_off_u32(&lsA[0][0]), gA, 32, 128, (unsigned)K);
        tdm_load_2d_bf16(lds_off_u32(&lsB[0][0]), gB, 32, 128, (unsigned)K);
    }
#else
    const int fr = tid >> 1;            // staging fallback: row 0..127
    const int fc = (tid & 1) * 16;      // half-row base
#endif

    for (int s = 0; s < steps; ++s) {
        const int buf = s & 1;
#ifdef HAVE_TDM
        if (wave == 0) {
            if (s + 1 < steps) {
                const int k1 = (s + 1) * 32;
                tdm_load_2d_bf16(lds_off_u32(&lsA[buf ^ 1][0]), gA + k1, 32, 128, (unsigned)K);
                tdm_load_2d_bf16(lds_off_u32(&lsB[buf ^ 1][0]), gB + k1, 32, 128, (unsigned)K);
                __builtin_amdgcn_s_wait_tensorcnt(2);   // stage s complete
            } else {
                __builtin_amdgcn_s_wait_tensorcnt(0);
            }
        }
#else
        {
            const int k0 = s * 32;
            v8bf a0 = *(const v8bf*)(gA + (size_t)fr * K + k0 + fc);
            v8bf a1 = *(const v8bf*)(gA + (size_t)fr * K + k0 + fc + 8);
            v8bf b0 = *(const v8bf*)(gB + (size_t)fr * K + k0 + fc);
            v8bf b1 = *(const v8bf*)(gB + (size_t)fr * K + k0 + fc + 8);
            *(v8bf*)&lsA[buf][fr * LDT + fc]     = a0;
            *(v8bf*)&lsA[buf][fr * LDT + fc + 8] = a1;
            *(v8bf*)&lsB[buf][fr * LDT + fc]     = b0;
            *(v8bf*)&lsB[buf][fr * LDT + fc + 8] = b1;
        }
#endif
        __syncthreads();

        v16bf afrag[4], bfrag[2];
#pragma unroll
        for (int mt = 0; mt < 4; ++mt)
            afrag[mt] = load_a_frag(&lsA[buf][(wm * 64 + mt * 16) * LDT], LDT, lane);
#pragma unroll
        for (int nt = 0; nt < 2; ++nt)
            bfrag[nt] = load_b_frag(&lsB[buf][(wn * 32 + nt * 16) * LDT], LDT, lane);

#pragma unroll
        for (int mt = 0; mt < 4; ++mt)
#pragma unroll
            for (int nt = 0; nt < 2; ++nt)
                acc[mt][nt] = wmma_bf16(afrag[mt], bfrag[nt], acc[mt][nt]);
        __syncthreads();
    }

    const int colc = lane & 15;
    const int rowo = (lane >> 4) * 8;
#pragma unroll
    for (int mt = 0; mt < 4; ++mt)
#pragma unroll
        for (int nt = 0; nt < 2; ++nt)
#pragma unroll
            for (int r = 0; r < 8; ++r) {
                const int row = m0 + wm * 64 + mt * 16 + rowo + r;
                const int col = n0 + wn * 32 + nt * 16 + colc;
                Out[(size_t)row * N + col] = acc[mt][nt][r];
            }
}

// ---------------------------------------------------------------------------
// RoPE (interleaved pairs), in place on [B*S, nheads*HD] fp32 buffer.
// ---------------------------------------------------------------------------
__global__ void rope_kernel(float* __restrict__ buf,
                            const float* __restrict__ cosT,
                            const float* __restrict__ sinT,
                            const int* __restrict__ pos_ids, int nheads) {
    const int hd2   = HD_ / 2;
    const int per_m = nheads * hd2;
    const long long idx = (long long)blockIdx.x * blockDim.x + threadIdx.x;
    if (idx >= (long long)BB * SS * per_m) return;
    const int m    = (int)(idx / per_m);
    const int rem  = (int)(idx - (long long)m * per_m);
    const int hidx = rem / hd2;
    const int i    = rem - hidx * hd2;
    const int pos  = pos_ids[m];
    const float c  = cosT[pos * hd2 + i];
    const float s  = sinT[pos * hd2 + i];
    const size_t base = (size_t)m * nheads * HD_ + (size_t)hidx * HD_ + 2 * i;
    const float x1 = buf[base], x2 = buf[base + 1];
    buf[base]     = x1 * c - x2 * s;
    buf[base + 1] = x1 * s + x2 * c;
}

// ---------------------------------------------------------------------------
// Flash attention (causal, GQA). 8 waves/block, wave = 16 q rows, CTA = 128
// rows of one head; KV streamed in 64-chunks via shared LDS tiles.
// Output written in bf16 (consumed by the bf16 O-projection GEMM).
// ---------------------------------------------------------------------------
#define LDK 136
#define LDV 72
#define LDP 72

__global__ __launch_bounds__(256) void flash_attn_kernel(const float* __restrict__ Q,
                                                         const float* __restrict__ K,
                                                         const float* __restrict__ V,
                                                         __bf16* __restrict__ Oout) {
    __shared__ __align__(16) __bf16 lsK[64 * LDK];      // [kv][hd]
    __shared__ __align__(16) __bf16 lsV[128 * LDV];     // [hd][kv] (V^T)
    __shared__ __align__(16) __bf16 lsP[8 * 16 * LDP];  // per-wave P

    const int tid  = threadIdx.x;
    const int lane = tid & 31;
    const int wave = tid >> 5;
    const int bh   = blockIdx.y;
    const int b    = bh / NH_;
    const int h    = bh % NH_;
    const int kvh  = h / GR_;
    const int q0   = blockIdx.x * 128;
    const int qw   = q0 + wave * 16;
    const int rowa = lane & 15;
    const int hb   = lane >> 4;
    const float qscale = 0.08838834764831845f;   // 1/sqrt(128)

    v16bf qf[4];
    const float* qbase = Q + (size_t)(b * SS + qw + rowa) * (NH_ * HD_) + h * HD_;
#pragma unroll
    for (int f = 0; f < 4; ++f) {
        const float* p = qbase + f * 32 + hb * 8;
        v4f v0 = *(const v4f*)(p);
        v4f v1 = *(const v4f*)(p + 4);
        v4f v2 = *(const v4f*)(p + 16);
        v4f v3 = *(const v4f*)(p + 20);
        v16bf a;
#pragma unroll
        for (int j = 0; j < 4; ++j) {
            a[j]      = (__bf16)(v0[j] * qscale);
            a[j + 4]  = (__bf16)(v1[j] * qscale);
            a[j + 8]  = (__bf16)(v2[j] * qscale);
            a[j + 12] = (__bf16)(v3[j] * qscale);
        }
        qf[f] = a;
    }

    float mrow[8], lrow[8];
    v8f oacc[8];
    const v8f zf = {0.f, 0.f, 0.f, 0.f, 0.f, 0.f, 0.f, 0.f};
#pragma unroll
    for (int r = 0; r < 8; ++r) { mrow[r] = -1e30f; lrow[r] = 0.f; }
#pragma unroll
    for (int dt = 0; dt < 8; ++dt) oacc[dt] = zf;

    const int nkb = (q0 + 128) / 64;
    const float* kgbase = K + (size_t)(b * SS) * (NKV_ * HD_) + kvh * HD_;
    const float* vgbase = V + (size_t)(b * SS) * (NKV_ * HD_) + kvh * HD_;

    const int lr = tid >> 2;
    const int lc = (tid & 3) * 32;

    for (int kb = 0; kb < nkb; ++kb) {
        const int kv0 = kb * 64;
        const float* gk = kgbase + (size_t)(kv0 + lr) * (NKV_ * HD_) + lc;
        const float* gv = vgbase + (size_t)(kv0 + lr) * (NKV_ * HD_) + lc;
        if (kb + 1 < nkb) {
            __builtin_prefetch(gk + (size_t)64 * (NKV_ * HD_), 0, 1);
            __builtin_prefetch(gv + (size_t)64 * (NKV_ * HD_), 0, 1);
        }
#pragma unroll
        for (int i = 0; i < 8; ++i) {
            v4f vk = *(const v4f*)(gk + 4 * i);
            v4bf hk;
#pragma unroll
            for (int j = 0; j < 4; ++j) hk[j] = (__bf16)vk[j];
            *(v4bf*)&lsK[lr * LDK + lc + 4 * i] = hk;
            v4f vv = *(const v4f*)(gv + 4 * i);
#pragma unroll
            for (int j = 0; j < 4; ++j)
                lsV[(lc + 4 * i + j) * LDV + lr] = (__bf16)vv[j];
        }
        __syncthreads();

        if (kv0 <= qw + 15) {   // wave-uniform causal skip
            v8f sacc[4];
#pragma unroll
            for (int nt = 0; nt < 4; ++nt) {
                sacc[nt] = zf;
#pragma unroll
                for (int kc = 0; kc < 4; ++kc)
                    sacc[nt] = wmma_bf16(
                        qf[kc],
                        load_b_frag(lsK + nt * 16 * LDK + kc * 32, LDK, lane),
                        sacc[nt]);
            }

#pragma unroll
            for (int r = 0; r < 8; ++r) {
                const int qrow = qw + r + hb * 8;
                float pm = -1e30f;
#pragma unroll
                for (int nt = 0; nt < 4; ++nt) {
                    const int col = kv0 + nt * 16 + rowa;
                    if (col > qrow) sacc[nt][r] = -1e30f;
                    pm = fmaxf(pm, sacc[nt][r]);
                }
                pm = fmaxf(pm, __shfl_xor(pm, 1, 32));
                pm = fmaxf(pm, __shfl_xor(pm, 2, 32));
                pm = fmaxf(pm, __shfl_xor(pm, 4, 32));
                pm = fmaxf(pm, __shfl_xor(pm, 8, 32));
                const float mnew  = fmaxf(mrow[r], pm);
                const float scale = __expf(mrow[r] - mnew);
                float ps = 0.f;
#pragma unroll
                for (int nt = 0; nt < 4; ++nt) {
                    const float e = __expf(sacc[nt][r] - mnew);
                    sacc[nt][r] = e;
                    ps += e;
                }
                ps += __shfl_xor(ps, 1, 32);
                ps += __shfl_xor(ps, 2, 32);
                ps += __shfl_xor(ps, 4, 32);
                ps += __shfl_xor(ps, 8, 32);
                lrow[r] = lrow[r] * scale + ps;
                mrow[r] = mnew;
#pragma unroll
                for (int dt = 0; dt < 8; ++dt) oacc[dt][r] *= scale;
            }

            __bf16* pw = lsP + wave * 16 * LDP;
#pragma unroll
            for (int nt = 0; nt < 4; ++nt)
#pragma unroll
                for (int r = 0; r < 8; ++r)
                    pw[(r + hb * 8) * LDP + nt * 16 + rowa] = (__bf16)sacc[nt][r];

            v16bf pa0 = load_a_frag(pw, LDP, lane);
            v16bf pa1 = load_a_frag(pw + 32, LDP, lane);

#pragma unroll
            for (int dt = 0; dt < 8; ++dt) {
                oacc[dt] = wmma_bf16(pa0, load_b_frag(lsV + dt * 16 * LDV, LDV, lane),
                                     oacc[dt]);
                oacc[dt] = wmma_bf16(pa1,
                                     load_b_frag(lsV + dt * 16 * LDV + 32, LDV, lane),
                                     oacc[dt]);
            }
        }
        __syncthreads();
    }

#pragma unroll
    for (int r = 0; r < 8; ++r) {
        const float inv = 1.0f / lrow[r];
        const size_t orow = (size_t)(b * SS + qw + r + hb * 8) * (NH_ * HD_) + h * HD_;
#pragma unroll
        for (int dt = 0; dt < 8; ++dt)
            Oout[orow + dt * 16 + rowa] = (__bf16)(oacc[dt][r] * inv);
    }
}

// ---------------------------------------------------------------------------
// Launch: bf16 pre-convert -> QKV projections -> RoPE(Q,K) -> flash attention
// -> O projection.   Workspace ~240MB.
// ---------------------------------------------------------------------------
extern "C" void kernel_launch(void* const* d_in, const int* in_sizes, int n_in,
                              void* d_out, int out_size, void* d_ws, size_t ws_size,
                              hipStream_t stream) {
    const float* hidden = (const float*)d_in[0];
    const float* cosT   = (const float*)d_in[1];
    const float* sinT   = (const float*)d_in[2];
    const int*   posid  = (const int*)d_in[3];
    // d_in[4] = attention_mask: causal, applied analytically in-kernel
    const float* Wq = (const float*)d_in[5];
    const float* Wk = (const float*)d_in[6];
    const float* Wv = (const float*)d_in[7];
    const float* Wo = (const float*)d_in[8];
    float* out = (float*)d_out;

    char* ws = (char*)d_ws;
    const size_t SZ_XB  = (size_t)4096 * 4096 * 2;   // bf16
    const size_t SZ_WKB = (size_t)1024 * 4096 * 2;   // bf16
    const size_t SZ_QF  = (size_t)4096 * 4096 * 4;   // fp32
    const size_t SZ_KF  = (size_t)4096 * 1024 * 4;   // fp32
    __bf16* Xb   = (__bf16*)(ws);                 ws += SZ_XB;
    __bf16* Wqb  = (__bf16*)(ws);                 ws += SZ_XB;
    __bf16* Wkb  = (__bf16*)(ws);                 ws += SZ_WKB;
    __bf16* Wvb  = (__bf16*)(ws);                 ws += SZ_WKB;
    __bf16* Wob  = (__bf16*)(ws);                 ws += SZ_XB;
    float*  qbuf = (float*)(ws);                  ws += SZ_QF;
    float*  kbuf = (float*)(ws);                  ws += SZ_KF;
    float*  vbuf = (float*)(ws);                  ws += SZ_KF;
    __bf16* abufb = (__bf16*)(ws);

    const dim3 blk(256);
    // one-time fp32 -> bf16 conversion of activations + weights
    const long long nX = (long long)4096 * 4096, nW = (long long)1024 * 4096;
    f32_to_bf16_kernel<<<(unsigned)(nX / 8 / 256), 256, 0, stream>>>(hidden, Xb, nX);
    f32_to_bf16_kernel<<<(unsigned)(nX / 8 / 256), 256, 0, stream>>>(Wq, Wqb, nX);
    f32_to_bf16_kernel<<<(unsigned)(nW / 8 / 256), 256, 0, stream>>>(Wk, Wkb, nW);
    f32_to_bf16_kernel<<<(unsigned)(nW / 8 / 256), 256, 0, stream>>>(Wv, Wvb, nW);
    f32_to_bf16_kernel<<<(unsigned)(nX / 8 / 256), 256, 0, stream>>>(Wo, Wob, nX);

    // Q/K/V projections
    gemm_xwt_kernel<<<dim3(32, 32), blk, 0, stream>>>(Xb, Wqb, qbuf, 4096, 4096, 4096);
    gemm_xwt_kernel<<<dim3(32, 8),  blk, 0, stream>>>(Xb, Wkb, kbuf, 4096, 1024, 4096);
    gemm_xwt_kernel<<<dim3(32, 8),  blk, 0, stream>>>(Xb, Wvb, vbuf, 4096, 1024, 4096);

    // RoPE on Q and K
    const long long nq = (long long)BB * SS * NH_ * (HD_ / 2);
    const long long nk = (long long)BB * SS * NKV_ * (HD_ / 2);
    rope_kernel<<<(unsigned)((nq + 255) / 256), 256, 0, stream>>>(qbuf, cosT, sinT, posid, NH_);
    rope_kernel<<<(unsigned)((nk + 255) / 256), 256, 0, stream>>>(kbuf, cosT, sinT, posid, NKV_);

    // Flash attention
    flash_attn_kernel<<<dim3(16, 64), blk, 0, stream>>>(qbuf, kbuf, vbuf, abufb);

    // Output projection (bf16 attn-out x bf16 Wo -> fp32)
    gemm_xwt_kernel<<<dim3(32, 32), blk, 0, stream>>>(abufb, Wob, out, 4096, 4096, 4096);
}
